// TwoWayAttention_42228118454501
// MI455X (gfx1250) — matrix operand order repaired and checked by reading, not compiled, not run on hardware
//
#include <hip/hip_runtime.h>
#include <hip/hip_bf16.h>

// ---------------------------------------------------------------------------
// TwoWayAttention block for MI455X (gfx1250): all GEMMs via
// v_wmma_f32_16x16x32_bf16 (bf16 inputs, f32 accumulate).
// ---------------------------------------------------------------------------

typedef __attribute__((ext_vector_type(16))) __bf16 v16bf;
typedef __attribute__((ext_vector_type(8)))  float  v8f;

#define FB_BT   1   // B operand stored N x K row-major (i.e. use B^T)
#define FB_RELU 2
#define FB_BIAS 4
#define FB_RES  8

// ---------------------------------------------------------------------------
// Fragment loaders. All loads are UNCONDITIONAL (no divergent branches around
// VMEM): the GUARD path clamps the address and zeroes the value with selects.
// A fragment: 16-bit A-matrix 16x32 layout — lane group 0 holds K {0..7,16..23},
// lane group 1 holds K {8..15,24..31}; VGPR v holds 2 consecutive K.
// ---------------------------------------------------------------------------
template<bool GUARD>
__device__ __forceinline__ v16bf load_a_frag(const float* __restrict__ Arow,
                                             int kb, int hi, int K)
{
    v16bf a;
    #pragma unroll
    for (int v = 0; v < 8; ++v) {
        int k0 = kb + ((v >> 2) << 4) + (hi << 3) + ((v & 3) << 1);
        float2 p;
        if (GUARD) {
            p = *(const float2*)(Arow + (k0 < K ? k0 : 0));   // safe addr, uncond load
            if (k0 >= K) { p.x = 0.0f; p.y = 0.0f; }          // v_cndmask, no branch
        } else {
            p = *(const float2*)(Arow + k0);                  // global_load_b64
        }
        a[2 * v]     = (__bf16)p.x;
        a[2 * v + 1] = (__bf16)p.y;
    }
    return a;
}

// B fragment: 32x16 (KxN) — lane group 0 holds K {0..7,16..23}, group 1 K {8..15,24..31};
// VGPR v holds 2 consecutive K for column n = lane&15.
template<bool BT, bool GUARD>
__device__ __forceinline__ v16bf load_b_frag(const float* __restrict__ Bbase,
                                             int kb, int hi, int K, int ldb)
{
    v16bf b;
    #pragma unroll
    for (int v = 0; v < 8; ++v) {
        int ka = kb + ((v >> 2) << 4) + (hi << 3) + ((v & 3) << 1);
        float y0, y1;
        if (BT) {
            // contiguous along K (B stored NxK): vectorized load
            float2 p = *(const float2*)(Bbase + (GUARD ? (ka < K ? ka : 0) : ka));
            y0 = p.x; y1 = p.y;
            if (GUARD && ka >= K) { y0 = 0.0f; y1 = 0.0f; }   // K even => pair-granular
        } else {
            int ka0 = GUARD ? (ka     < K ? ka     : 0) : ka;
            int ka1 = GUARD ? (ka + 1 < K ? ka + 1 : 0) : ka + 1;
            y0 = Bbase[(long)ka0 * ldb];
            y1 = Bbase[(long)ka1 * ldb];
            if (GUARD) {
                if (ka     >= K) y0 = 0.0f;
                if (ka + 1 >= K) y1 = 0.0f;
            }
        }
        b[2 * v]     = (__bf16)y0;
        b[2 * v + 1] = (__bf16)y1;
    }
    return b;
}

// D layout: VGPR v -> row v + 8*hi, col = lane&15.
__device__ __forceinline__ void store_tile(float* __restrict__ Cb,
                                           const float* __restrict__ Rb,
                                           const float* __restrict__ bias,
                                           v8f acc, int row0, int col,
                                           int ldc, int hi, int flags)
{
    float bv = (flags & FB_BIAS) ? bias[col] : 0.0f;
    #pragma unroll
    for (int v = 0; v < 8; ++v) {
        int row = row0 + v + (hi << 3);
        float val = acc[v] + bv;
        if (flags & FB_RES)  val += Rb[(long)row * ldc + col];
        if (flags & FB_RELU) val = fmaxf(val, 0.0f);
        Cb[(long)row * ldc + col] = val;
    }
}

#define WMMA_BF16(A_, B_, C_) \
    __builtin_amdgcn_wmma_f32_16x16x32_bf16(false, (A_), false, (B_), (short)0, (C_), false, false)

// ---------------------------------------------------------------------------
// Generic batched GEMM:  C[z] = act( A[z] (MxK) * B[z] (KxN) + bias + Res[z] )
// z (blockIdx.y) decomposes as zo=z/nInner, zi=z%nInner; operand offset =
// zo*Out + zi*In (addresses per-(batch,head) slices). 4 waves per block.
// ---------------------------------------------------------------------------

// 16x16 tile per wave (fallback: K%32!=0 or N/M not multiples of 32)
template<bool BT, bool GUARD>
__global__ void __launch_bounds__(128)
gemm16_kern(const float* __restrict__ A, const float* __restrict__ Bm,
            const float* __restrict__ bias, const float* __restrict__ Res,
            float* __restrict__ C,
            int M, int N, int K, int lda, int ldb, int ldc, int nInner,
            long aOut, long aIn, long bOut, long bIn,
            long cOut, long cIn, long rOut, long rIn, int flags)
{
    const int wave = threadIdx.x >> 5;
    const int lane = threadIdx.x & 31;
    const int tilesN = N >> 4;
    const int tilesTotal = (M >> 4) * tilesN;
    const int tile = blockIdx.x * 4 + wave;           // wave-uniform
    if (tile >= tilesTotal) return;                   // wave-uniform exit
    const int tm = tile / tilesN;
    const int tn = tile % tilesN;

    const int z  = blockIdx.y;
    const int zo = z / nInner;
    const int zi = z % nInner;
    const float* Ab = A  + (long)zo * aOut + (long)zi * aIn;
    const float* Bb = Bm + (long)zo * bOut + (long)zi * bIn;
    float*       Cb = C  + (long)zo * cOut + (long)zi * cIn;
    const float* Rb = (flags & FB_RES) ? (Res + (long)zo * rOut + (long)zi * rIn) : nullptr;

    const int mi = lane & 15;
    const int hi = lane >> 4;

    const float* Arow  = Ab + (long)(tm * 16 + mi) * lda;
    const float* Bbase = BT ? (Bb + (long)(tn * 16 + mi) * ldb) : (Bb + (tn * 16 + mi));

    v8f acc = {};
    for (int kb = 0; kb < K; kb += 32) {
        v16bf a = load_a_frag<GUARD>(Arow, kb, hi, K);
        v16bf b = load_b_frag<BT, GUARD>(Bbase, kb, hi, K, ldb);
        acc = WMMA_BF16(a, b, acc);
    }
    store_tile(Cb, Rb, bias, acc, tm * 16, tn * 16 + mi, ldc, hi, flags);
}

// 32x32 tile per wave: 2 A frags x 2 B frags -> 4 WMMAs per k-step (2x reuse).
template<bool BT>
__global__ void __launch_bounds__(128)
gemm32_kern(const float* __restrict__ A, const float* __restrict__ Bm,
            const float* __restrict__ bias, const float* __restrict__ Res,
            float* __restrict__ C,
            int M, int N, int K, int lda, int ldb, int ldc, int nInner,
            long aOut, long aIn, long bOut, long bIn,
            long cOut, long cIn, long rOut, long rIn, int flags)
{
    const int wave = threadIdx.x >> 5;
    const int lane = threadIdx.x & 31;
    const int tilesN = N >> 5;
    const int tilesTotal = (M >> 5) * tilesN;
    const int tile = blockIdx.x * 4 + wave;           // wave-uniform
    if (tile >= tilesTotal) return;                   // wave-uniform exit
    const int tm = tile / tilesN;
    const int tn = tile % tilesN;

    const int z  = blockIdx.y;
    const int zo = z / nInner;
    const int zi = z % nInner;
    const float* Ab = A  + (long)zo * aOut + (long)zi * aIn;
    const float* Bb = Bm + (long)zo * bOut + (long)zi * bIn;
    float*       Cb = C  + (long)zo * cOut + (long)zi * cIn;
    const float* Rb = (flags & FB_RES) ? (Res + (long)zo * rOut + (long)zi * rIn) : nullptr;

    const int mi = lane & 15;
    const int hi = lane >> 4;

    const float* Arow0 = Ab + (long)(tm * 32 + mi) * lda;
    const float* Arow1 = Arow0 + (long)16 * lda;
    const float* Bb0   = BT ? (Bb + (long)(tn * 32 + mi) * ldb) : (Bb + (tn * 32 + mi));
    const float* Bb1   = BT ? (Bb0 + (long)16 * ldb)            : (Bb0 + 16);

    v8f acc00 = {}, acc01 = {}, acc10 = {}, acc11 = {};
    for (int kb = 0; kb < K; kb += 32) {
        v16bf a0 = load_a_frag<false>(Arow0, kb, hi, K);
        v16bf a1 = load_a_frag<false>(Arow1, kb, hi, K);
        v16bf b0 = load_b_frag<BT, false>(Bb0, kb, hi, K, ldb);
        v16bf b1 = load_b_frag<BT, false>(Bb1, kb, hi, K, ldb);
        acc00 = WMMA_BF16(a0, b0, acc00);
        acc01 = WMMA_BF16(a0, b1, acc01);
        acc10 = WMMA_BF16(a1, b0, acc10);
        acc11 = WMMA_BF16(a1, b1, acc11);
    }
    store_tile(Cb, Rb, bias, acc00, tm * 32,      tn * 32 + mi,      ldc, hi, flags);
    store_tile(Cb, Rb, bias, acc01, tm * 32,      tn * 32 + 16 + mi, ldc, hi, flags);
    store_tile(Cb, Rb, bias, acc10, tm * 32 + 16, tn * 32 + mi,      ldc, hi, flags);
    store_tile(Cb, Rb, bias, acc11, tm * 32 + 16, tn * 32 + 16 + mi, ldc, hi, flags);
}

// ---------------------------------------------------------------------------
// Elementwise add: c = a + b
// ---------------------------------------------------------------------------
__global__ void add_kernel(const float* __restrict__ a, const float* __restrict__ b,
                           float* __restrict__ c, long n)
{
    long i = (long)blockIdx.x * blockDim.x + threadIdx.x;
    if (i < n) c[i] = a[i] + b[i];
}

// ---------------------------------------------------------------------------
// Row softmax of s*scale, in place. One 256-thread block per row of length L.
// ---------------------------------------------------------------------------
__global__ void softmax_kernel(float* __restrict__ s, int L, float scale)
{
    __shared__ float red[256];
    const long base = (long)blockIdx.x * L;
    const int  t    = threadIdx.x;

    float mx = -3.402823466e38f;
    for (int i = t; i < L; i += 256) mx = fmaxf(mx, s[base + i] * scale);
    red[t] = mx; __syncthreads();
    for (int st = 128; st > 0; st >>= 1) {
        if (t < st) red[t] = fmaxf(red[t], red[t + st]);
        __syncthreads();
    }
    mx = red[0]; __syncthreads();

    float sum = 0.0f;
    for (int i = t; i < L; i += 256) {
        float e = __expf(s[base + i] * scale - mx);
        s[base + i] = e;
        sum += e;
    }
    red[t] = sum; __syncthreads();
    for (int st = 128; st > 0; st >>= 1) {
        if (t < st) red[t] += red[t + st];
        __syncthreads();
    }
    float inv = 1.0f / red[0];
    for (int i = t; i < L; i += 256) s[base + i] *= inv;
}

// ---------------------------------------------------------------------------
// LayerNorm over last dim = 256, eps = 1e-3 (keras default in reference).
// One 256-thread block per row.
// ---------------------------------------------------------------------------
__global__ void ln_kernel(const float* __restrict__ x, const float* __restrict__ g,
                          const float* __restrict__ bb, float* __restrict__ y)
{
    __shared__ float red[256];
    const int  t   = threadIdx.x;
    const long row = blockIdx.x;

    float v = x[row * 256 + t];
    red[t] = v; __syncthreads();
    for (int s = 128; s > 0; s >>= 1) {
        if (t < s) red[t] += red[t + s];
        __syncthreads();
    }
    float mean = red[0] * (1.0f / 256.0f); __syncthreads();

    float d = v - mean;
    red[t] = d * d; __syncthreads();
    for (int s = 128; s > 0; s >>= 1) {
        if (t < s) red[t] += red[t + s];
        __syncthreads();
    }
    float var = red[0] * (1.0f / 256.0f);
    y[row * 256 + t] = d * rsqrtf(var + 1e-3f) * g[t] + bb[t];
}

// ---------------------------------------------------------------------------
// Host-side GEMM dispatcher
// ---------------------------------------------------------------------------
static void gemm(hipStream_t s, const float* A, const float* B, const float* bias,
                 const float* Res, float* C, int M, int N, int K,
                 int lda, int ldb, int ldc,
                 int batch, int nInner,
                 long aOut, long aIn, long bOut, long bIn,
                 long cOut, long cIn, long rOut, long rIn, int flags)
{
    const bool bt    = (flags & FB_BT) != 0;
    const bool guard = (K & 31) != 0;
    dim3 block(128, 1, 1);
    if (!guard && (M % 32 == 0) && (N % 32 == 0)) {
        int tiles = (M / 32) * (N / 32);
        dim3 grid((tiles + 3) / 4, batch, 1);
        if (bt) gemm32_kern<true ><<<grid, block, 0, s>>>(A, B, bias, Res, C, M, N, K, lda, ldb, ldc,
                                                          nInner, aOut, aIn, bOut, bIn, cOut, cIn, rOut, rIn, flags);
        else    gemm32_kern<false><<<grid, block, 0, s>>>(A, B, bias, Res, C, M, N, K, lda, ldb, ldc,
                                                          nInner, aOut, aIn, bOut, bIn, cOut, cIn, rOut, rIn, flags);
    } else {
        int tiles = (M / 16) * (N / 16);
        dim3 grid((tiles + 3) / 4, batch, 1);
        if (bt) {
            if (guard) gemm16_kern<true, true ><<<grid, block, 0, s>>>(A, B, bias, Res, C, M, N, K, lda, ldb, ldc,
                                                                       nInner, aOut, aIn, bOut, bIn, cOut, cIn, rOut, rIn, flags);
            else       gemm16_kern<true, false><<<grid, block, 0, s>>>(A, B, bias, Res, C, M, N, K, lda, ldb, ldc,
                                                                       nInner, aOut, aIn, bOut, bIn, cOut, cIn, rOut, rIn, flags);
        } else {
            if (guard) gemm16_kern<false, true ><<<grid, block, 0, s>>>(A, B, bias, Res, C, M, N, K, lda, ldb, ldc,
                                                                        nInner, aOut, aIn, bOut, bIn, cOut, cIn, rOut, rIn, flags);
            else       gemm16_kern<false, false><<<grid, block, 0, s>>>(A, B, bias, Res, C, M, N, K, lda, ldb, ldc,
                                                                        nInner, aOut, aIn, bOut, bIn, cOut, cIn, rOut, rIn, flags);
        }
    }
}

extern "C" void kernel_launch(void* const* d_in, const int* in_sizes, int n_in,
                              void* d_out, int out_size, void* d_ws, size_t ws_size,
                              hipStream_t stream)
{
    (void)in_sizes; (void)n_in; (void)out_size; (void)ws_size;

    // ---- model dims ----
    const int  B_ = 8, Nq = 256, Nk = 4096, D = 256, DS = 128, H = 8, MLPD = 2048;
    const long NqD = (long)Nq * D;    // 65536
    const long NqS = (long)Nq * DS;   // 32768
    const long NkS = (long)Nk * DS;   // 524288
    const int  Mq  = B_ * Nq;         // 2048
    const int  Mk  = B_ * Nk;         // 32768
    const float selfScale  = 0.17677669529663689f;  // 1/sqrt(32)
    const float crossScale = 0.25f;                 // 1/sqrt(16)

    // ---- inputs (setup_inputs dict order, nested dicts in insertion order) ----
    const float* queries  = (const float*)d_in[0];
    const float* keys     = (const float*)d_in[1];
    const float* query_pe = (const float*)d_in[2];
    const float* key_pe   = (const float*)d_in[3];
    const float* sa_wq = (const float*)d_in[4],  *sa_bq = (const float*)d_in[5];
    const float* sa_wk = (const float*)d_in[6],  *sa_bk = (const float*)d_in[7];
    const float* sa_wv = (const float*)d_in[8],  *sa_bv = (const float*)d_in[9];
    const float* sa_wo = (const float*)d_in[10], *sa_bo = (const float*)d_in[11];
    const float* t2_wq = (const float*)d_in[12], *t2_bq = (const float*)d_in[13];
    const float* t2_wk = (const float*)d_in[14], *t2_bk = (const float*)d_in[15];
    const float* t2_wv = (const float*)d_in[16], *t2_bv = (const float*)d_in[17];
    const float* t2_wo = (const float*)d_in[18], *t2_bo = (const float*)d_in[19];
    const float* i2_wq = (const float*)d_in[20], *i2_bq = (const float*)d_in[21];
    const float* i2_wk = (const float*)d_in[22], *i2_bk = (const float*)d_in[23];
    const float* i2_wv = (const float*)d_in[24], *i2_bv = (const float*)d_in[25];
    const float* i2_wo = (const float*)d_in[26], *i2_bo = (const float*)d_in[27];
    const float* m_w1  = (const float*)d_in[28], *m_b1  = (const float*)d_in[29];
    const float* m_w2  = (const float*)d_in[30], *m_b2  = (const float*)d_in[31];
    const float* g1 = (const float*)d_in[32], *lb1 = (const float*)d_in[33];
    const float* g2 = (const float*)d_in[34], *lb2 = (const float*)d_in[35];
    const float* g3 = (const float*)d_in[36], *lb3 = (const float*)d_in[37];
    const float* g4 = (const float*)d_in[38], *lb4 = (const float*)d_in[39];

    // ---- workspace layout (floats) : ~176 MB total ----
    float* ws   = (float*)d_ws;
    float* qpe  = ws;                  // [8,256,256]   524288
    float* kpe  = qpe  + 524288;       // [8,4096,256]  8388608
    float* pq   = kpe  + 8388608;      // [8,256,256]   524288
    float* pk   = pq   + 524288;       // [8,4096,128]  4194304 (also self-k 2048x256)
    float* pv   = pk   + 4194304;      // [8,4096,128]  4194304
    float* obuf = pv   + 4194304;      // [8,4096,128]  4194304
    float* t0   = obuf + 4194304;      // [2048,256]    524288  (pre-LN queries)
    float* tbig = t0   + 524288;       // [32768,256]   8388608 (pre-LN keys)
    float* mlph = tbig + 8388608;      // [2048,2048]   4194304
    float* xq   = mlph + 4194304;      // [2048,256]    524288  (running queries)
    float* sc   = xq   + 524288;       // per-b scores  8388608

    float* qout = (float*)d_out;            // final queries [8,256,256]
    float* kout = qout + 524288;            // final keys    [8,4096,256]

    // ================= self attention =================
    add_kernel<<<(Mq * D + 255) / 256, 256, 0, stream>>>(queries, query_pe, qpe, (long)Mq * D);
    gemm(stream, qpe,     sa_wq, sa_bq, nullptr, pq, Mq, D, D, D, D, D, 1, 1, 0,0,0,0,0,0,0,0, FB_BIAS);
    gemm(stream, qpe,     sa_wk, sa_bk, nullptr, pk, Mq, D, D, D, D, D, 1, 1, 0,0,0,0,0,0,0,0, FB_BIAS);
    gemm(stream, queries, sa_wv, sa_bv, nullptr, pv, Mq, D, D, D, D, D, 1, 1, 0,0,0,0,0,0,0,0, FB_BIAS);
    // scores[b,h] = q_h @ k_h^T : batch 64, K=32
    gemm(stream, pq, pk, nullptr, nullptr, sc, Nq, Nq, 32, D, D, Nq,
         B_ * H, H, NqD, 32, NqD, 32, (long)H * Nq * Nq, (long)Nq * Nq, 0, 0, FB_BT);
    softmax_kernel<<<B_ * H * Nq, 256, 0, stream>>>(sc, Nq, selfScale);
    // o[b,h] = probs @ v_h
    gemm(stream, sc, pv, nullptr, nullptr, obuf, Nq, 32, Nq, Nq, D, D,
         B_ * H, H, (long)H * Nq * Nq, (long)Nq * Nq, NqD, 32, NqD, 32, 0, 0, 0);
    gemm(stream, obuf, sa_wo, sa_bo, queries, t0, Mq, D, D, D, D, D, 1, 1, 0,0,0,0,0,0,0,0, FB_BIAS | FB_RES);
    ln_kernel<<<Mq, 256, 0, stream>>>(t0, g1, lb1, xq);

    // ================= token -> image cross attention =================
    add_kernel<<<(Mq * D + 255) / 256, 256, 0, stream>>>(xq,   query_pe, qpe, (long)Mq * D);
    add_kernel<<<(Mk * D + 255) / 256, 256, 0, stream>>>(keys, key_pe,   kpe, (long)Mk * D);
    gemm(stream, qpe,  t2_wq, t2_bq, nullptr, pq, Mq, DS, D, D, DS, DS, 1, 1, 0,0,0,0,0,0,0,0, FB_BIAS);
    gemm(stream, kpe,  t2_wk, t2_bk, nullptr, pk, Mk, DS, D, D, DS, DS, 1, 1, 0,0,0,0,0,0,0,0, FB_BIAS);
    gemm(stream, keys, t2_wv, t2_bv, nullptr, pv, Mk, DS, D, D, DS, DS, 1, 1, 0,0,0,0,0,0,0,0, FB_BIAS);
    for (int b = 0; b < B_; ++b) {   // chunk by batch: scores are 8x256x4096 per b
        gemm(stream, pq + b * NqS, pk + b * NkS, nullptr, nullptr, sc,
             Nq, Nk, 16, DS, DS, Nk, H, H, 0, 16, 0, 16, 0, (long)Nq * Nk, 0, 0, FB_BT);
        softmax_kernel<<<H * Nq, 256, 0, stream>>>(sc, Nk, crossScale);
        gemm(stream, sc, pv + b * NkS, nullptr, nullptr, obuf + b * NqS,
             Nq, 16, Nk, Nk, DS, DS, H, H, 0, (long)Nq * Nk, 0, 16, 0, 16, 0, 0, 0);
    }
    gemm(stream, obuf, t2_wo, t2_bo, xq, t0, Mq, D, DS, DS, D, D, 1, 1, 0,0,0,0,0,0,0,0, FB_BIAS | FB_RES);
    ln_kernel<<<Mq, 256, 0, stream>>>(t0, g2, lb2, xq);

    // ================= MLP =================
    gemm(stream, xq,   m_w1, m_b1, nullptr, mlph, Mq, MLPD, D, D, MLPD, MLPD, 1, 1, 0,0,0,0,0,0,0,0, FB_BIAS | FB_RELU);
    gemm(stream, mlph, m_w2, m_b2, xq,      t0,   Mq, D, MLPD, MLPD, D, D,     1, 1, 0,0,0,0,0,0,0,0, FB_BIAS | FB_RES);
    ln_kernel<<<Mq, 256, 0, stream>>>(t0, g3, lb3, qout);   // final queries -> d_out

    // ================= image -> token cross attention =================
    add_kernel<<<(Mq * D + 255) / 256, 256, 0, stream>>>(qout, query_pe, qpe, (long)Mq * D);
    // kpe (= keys + key_pe) still valid from above
    gemm(stream, kpe,  i2_wq, i2_bq, nullptr, pk, Mk, DS, D, D, DS, DS, 1, 1, 0,0,0,0,0,0,0,0, FB_BIAS); // q (image side)
    gemm(stream, qpe,  i2_wk, i2_bk, nullptr, pq, Mq, DS, D, D, DS, DS, 1, 1, 0,0,0,0,0,0,0,0, FB_BIAS); // k (token side)
    gemm(stream, qout, i2_wv, i2_bv, nullptr, pv, Mq, DS, D, D, DS, DS, 1, 1, 0,0,0,0,0,0,0,0, FB_BIAS); // v (token side)
    for (int b = 0; b < B_; ++b) {   // scores are 8x4096x256 per b
        gemm(stream, pk + b * NkS, pq + b * NqS, nullptr, nullptr, sc,
             Nk, Nq, 16, DS, DS, Nq, H, H, 0, 16, 0, 16, 0, (long)Nk * Nq, 0, 0, FB_BT);
        softmax_kernel<<<H * Nk, 256, 0, stream>>>(sc, Nq, crossScale);
        gemm(stream, sc, pv + b * NqS, nullptr, nullptr, obuf + b * NkS,
             Nk, 16, Nq, Nq, DS, DS, H, H, 0, (long)Nk * Nq, 0, 16, 0, 16, 0, 0, 0);
    }
    gemm(stream, obuf, i2_wo, i2_bo, keys, tbig, Mk, D, DS, DS, D, D, 1, 1, 0,0,0,0,0,0,0,0, FB_BIAS | FB_RES);
    ln_kernel<<<Mk, 256, 0, stream>>>(tbig, g4, lb4, kout);  // final keys -> d_out + 524288
}
